// RAMTransformerLM_60644938219956
// MI455X (gfx1250) — matrix operand order repaired and checked by reading, not compiled, not run on hardware
//
#include <hip/hip_runtime.h>
#include <hip/hip_bf16.h>

// ---------------------------------------------------------------------------
// RAM-transformer forward for MI455X (gfx1250, wave32, WMMA).
// All GEMMs use v_wmma_f32_16x16x32_f16 (f16 in / f32 accumulate).
// LDS staging uses CDNA5 async global->LDS copies (ASYNCcnt) where no
// transpose is needed; cross-attention collapses to a per-layer constant
// vector because mem == 0 in the reference.
// ---------------------------------------------------------------------------

typedef __attribute__((ext_vector_type(16))) _Float16 v16h;
typedef __attribute__((ext_vector_type(8)))  float    v8f;

#define CB   8
#define CS   512
#define CV   32000
#define CCTX 8
#define CNC  512
#define CD   256
#define CNH  4
#define CL   4
#define CFF  512
#define CROWS (CB*CS)   // 4096 token rows

// Async memory->LDS copy, 16 bytes per lane, tracked on ASYNCcnt.
// Flat pointers to __shared__ have LDS offset in the low 32 bits (ISA 10.2),
// so truncation yields the VDST LDS address.
__device__ __forceinline__ void async_copy_b128(_Float16* lds, const _Float16* g)
{
  unsigned loff = (unsigned)(unsigned long long)lds;
  asm volatile("global_load_async_to_lds_b128 %0, %1, off"
               :: "v"(loff), "v"(g) : "memory");
}
__device__ __forceinline__ void wait_async0()
{
  asm volatile("s_wait_asynccnt 0x0" ::: "memory");
}

// ---------------------------------------------------------------------------
// Generic tiled WMMA GEMM: C[M,N] = act( scale*(A[M,K] @ B) + bias + residual )
// A: f16 row-major (lda). B: f16; transB=0 -> B is KxN (ldb), transB=1 -> B is
// NxK (ldb) i.e. we multiply by B^T. Batched over gridDim.z with (b,h) shift/
// mask split. Block = 128 threads = 4 waves; block tile 64x64; each wave owns
// a 32x32 tile = 2x2 v_wmma_f32_16x16x32_f16 frags.
// ---------------------------------------------------------------------------
__launch_bounds__(128)
__global__ void gemm_f16_wmma(
    const _Float16* __restrict__ A, int lda,
    const _Float16* __restrict__ Bm, int ldb, int transB,
    float* __restrict__ Cf, _Float16* __restrict__ Ch, int ldc,
    int M, int N, int K,
    const float* __restrict__ bias,
    const float* __restrict__ residual, int ldres, int resMask, // resMod-1 (pow2)
    float scale, int doRelu,
    int nhShift, int nhMask,
    long long sAb, long long sAh,
    long long sBb, long long sBh,
    long long sCb, long long sCh)
{
  __shared__ _Float16 As[64][40];   // [m][k], +8 halves pad
  __shared__ _Float16 Bs[64][40];   // [n][k] (B staged transposed)

  const int z  = blockIdx.z;
  const int zb = z >> nhShift;
  const int zh = z & nhMask;
  A  += zb * sAb + zh * sAh;
  Bm += zb * sBb + zh * sBh;
  const long long coff = zb * sCb + zh * sCh;

  const int tileN = blockIdx.x * 64;
  const int tileM = blockIdx.y * 64;

  const int t    = threadIdx.x;
  const int lane = t & 31;
  const int wave = t >> 5;
  const int wm   = wave >> 1;        // 0..1 : wave row
  const int wn   = wave & 1;         // 0..1 : wave col
  const int hi8  = (lane >> 4) << 3; // 0 or 8
  const int l16  = lane & 15;

  v8f acc[2][2] = {};

  for (int k0 = 0; k0 < K; k0 += 32) {
    // ---- stage A tile 64x32: async global->LDS (no VGPR round-trip)
    {
      const int r  = t >> 1;
      const int c0 = (t & 1) * 16;
      const _Float16* src = A + (long long)(tileM + r) * lda + (k0 + c0);
      async_copy_b128(&As[r][c0],     src);
      async_copy_b128(&As[r][c0 + 8], src + 8);
      if (k0 + 32 < K) __builtin_prefetch(src + 32, 0, 3);   // WGP-near prefetch
    }
    // ---- stage B tile into [n][k]
    if (transB) {  // B given as NxK row-major: direct async copy
      const int r  = t >> 1;
      const int c0 = (t & 1) * 16;
      const _Float16* src = Bm + (long long)(tileN + r) * ldb + (k0 + c0);
      async_copy_b128(&Bs[r][c0],     src);
      async_copy_b128(&Bs[r][c0 + 8], src + 8);
    } else {       // B given as KxN row-major: transpose while staging (VGPR path)
      const int k  = t >> 2;          // 0..31
      const int n0 = (t & 3) * 16;    // 0,16,32,48
      const _Float16* src = Bm + (long long)(k0 + k) * ldb + (tileN + n0);
      union { int4 q[2]; _Float16 h[16]; } tmp;
      const int4* s4 = reinterpret_cast<const int4*>(src);
      tmp.q[0] = s4[0];
      tmp.q[1] = s4[1];
      #pragma unroll
      for (int j = 0; j < 16; ++j) Bs[n0 + j][k] = tmp.h[j];
    }
    wait_async0();       // ASYNCcnt==0: our async LDS writes have landed
    __syncthreads();

    // ---- build fragments per the CDNA5 16-bit WMMA VGPR layouts
    union Frag { v16h v; unsigned u[8]; };
    Frag a0, a1, b0, b1;
    const int ar0 = wm * 32 + l16;
    const int ar1 = ar0 + 16;
    #pragma unroll
    for (int g = 0; g < 8; ++g) {
      // A 16x32: lanes0-15 K = {0..7,16..23} pairs, lanes16-31 +8
      const int kk = (g < 4) ? (hi8 + 2 * g) : (16 + hi8 + 2 * (g - 4));
      a0.u[g] = *reinterpret_cast<const unsigned*>(&As[ar0][kk]);
      a1.u[g] = *reinterpret_cast<const unsigned*>(&As[ar1][kk]);
    }
    const int bc0 = wn * 32 + l16;
    const int bc1 = bc0 + 16;
    #pragma unroll
    for (int g = 0; g < 8; ++g) {
      // B 32x16: same K order as A (lanes0-15 K = {0..7,16..23}, lanes16-31 +8)
      const int kb = (g < 4) ? (hi8 + 2 * g) : (16 + hi8 + 2 * (g - 4));
      b0.u[g] = *reinterpret_cast<const unsigned*>(&Bs[bc0][kb]);
      b1.u[g] = *reinterpret_cast<const unsigned*>(&Bs[bc1][kb]);
    }

    acc[0][0] = __builtin_amdgcn_wmma_f32_16x16x32_f16(false, a0.v, false, b0.v, (short)0, acc[0][0], false, false);
    acc[0][1] = __builtin_amdgcn_wmma_f32_16x16x32_f16(false, a0.v, false, b1.v, (short)0, acc[0][1], false, false);
    acc[1][0] = __builtin_amdgcn_wmma_f32_16x16x32_f16(false, a1.v, false, b0.v, (short)0, acc[1][0], false, false);
    acc[1][1] = __builtin_amdgcn_wmma_f32_16x16x32_f16(false, a1.v, false, b1.v, (short)0, acc[1][1], false, false);

    __syncthreads();
  }

  // ---- epilogue (C/D layout: VGPR r -> M = r + hi8, N = lane&15)
  float* cf = Cf ? Cf + coff : nullptr;
  _Float16* ch = Ch ? Ch + coff : nullptr;
  #pragma unroll
  for (int i = 0; i < 2; ++i) {
    #pragma unroll
    for (int j = 0; j < 2; ++j) {
      const int col = tileN + wn * 32 + j * 16 + l16;
      #pragma unroll
      for (int r = 0; r < 8; ++r) {
        const int row = tileM + wm * 32 + i * 16 + hi8 + r;
        float v = acc[i][j][r] * scale;
        if (bias)     v += bias[col];
        if (residual) v += residual[(long long)(row & resMask) * ldres + col];
        if (doRelu)   v = v > 0.f ? v : 0.f;
        if (cf) cf[(long long)row * ldc + col] = v;
        if (ch) ch[(long long)row * ldc + col] = (_Float16)v;
      }
    }
  }
}

// ---------------------------------------------------------------------------
// Embedding: scores[b,s,c] = mean_{j<8} ram_table[win(b,s,j), c]  -> f16
// ---------------------------------------------------------------------------
__launch_bounds__(256)
__global__ void embed_scores(const int* __restrict__ tok,
                             const float* __restrict__ ram,
                             _Float16* __restrict__ sc)
{
  const int bs = blockIdx.x;            // 0..4095
  const int b  = bs >> 9;
  const int s  = bs & (CS - 1);
  const int tid = threadIdx.x;
  int rows[CCTX];
  #pragma unroll
  for (int j = 0; j < CCTX; ++j) {
    const int ti = s + j - (CCTX - 1);
    rows[j] = (ti < 0) ? 0 : tok[b * CS + ti];   // PAD=0 -> ram_table row 0
  }
  #pragma unroll
  for (int c = tid; c < CNC; c += 256) {
    float a = 0.f;
    #pragma unroll
    for (int j = 0; j < CCTX; ++j) a += ram[(long long)rows[j] * CNC + c];
    sc[(long long)bs * CNC + c] = (_Float16)(a * 0.125f);
  }
}

// ---------------------------------------------------------------------------
// LayerNorm over D=256, f32 in -> f16 out (GEMM-ready)
// ---------------------------------------------------------------------------
__launch_bounds__(256)
__global__ void layernorm_f16(const float* __restrict__ x,
                              const float* __restrict__ s,
                              const float* __restrict__ b,
                              _Float16* __restrict__ h)
{
  __shared__ float red[256];
  const int row = blockIdx.x;
  const int tid = threadIdx.x;
  const float v = x[(long long)row * CD + tid];
  red[tid] = v; __syncthreads();
  #pragma unroll
  for (int o = 128; o > 0; o >>= 1) { if (tid < o) red[tid] += red[tid + o]; __syncthreads(); }
  const float m = red[0] * (1.f / CD);
  __syncthreads();
  const float d = v - m;
  red[tid] = d * d; __syncthreads();
  #pragma unroll
  for (int o = 128; o > 0; o >>= 1) { if (tid < o) red[tid] += red[tid + o]; __syncthreads(); }
  const float rs = rsqrtf(red[0] * (1.f / CD) + 1e-5f);
  h[(long long)row * CD + tid] = (_Float16)(d * rs * s[tid] + b[tid]);
}

// ---------------------------------------------------------------------------
// Causal softmax over rows of length S=512; f32 scores -> f16 probs (0 masked)
// grid: (S, B*NH)
// ---------------------------------------------------------------------------
__launch_bounds__(256)
__global__ void softmax_causal(const float* __restrict__ sc,
                               _Float16* __restrict__ p)
{
  __shared__ float red[256];
  const int q = blockIdx.x;
  const long long base = ((long long)blockIdx.y * CS + q) * CS;
  const int tid = threadIdx.x;
  const float v0 = (tid       <= q) ? sc[base + tid]       : -3.0e38f;
  const float v1 = (tid + 256 <= q) ? sc[base + tid + 256] : -3.0e38f;
  red[tid] = fmaxf(v0, v1); __syncthreads();
  #pragma unroll
  for (int o = 128; o > 0; o >>= 1) { if (tid < o) red[tid] = fmaxf(red[tid], red[tid + o]); __syncthreads(); }
  const float mx = red[0];
  __syncthreads();
  const float e0 = (tid       <= q) ? __expf(v0 - mx) : 0.f;
  const float e1 = (tid + 256 <= q) ? __expf(v1 - mx) : 0.f;
  red[tid] = e0 + e1; __syncthreads();
  #pragma unroll
  for (int o = 128; o > 0; o >>= 1) { if (tid < o) red[tid] += red[tid + o]; __syncthreads(); }
  const float inv = 1.f / red[0];
  p[base + tid]       = (_Float16)(e0 * inv);
  p[base + tid + 256] = (_Float16)(e1 * inv);
}

// ---------------------------------------------------------------------------
// f32 -> f16 conversion (weights, final activations)
// ---------------------------------------------------------------------------
__global__ void f32_to_f16(const float* __restrict__ src,
                           _Float16* __restrict__ dst, long long n)
{
  for (long long i = blockIdx.x * (long long)blockDim.x + threadIdx.x;
       i < n; i += (long long)gridDim.x * blockDim.x)
    dst[i] = (_Float16)src[i];
}

// ---------------------------------------------------------------------------
// Cross-attention constant: cvec[l,d] = sum_e ca_bv[l,e]*ca_Wo[l,e,d] + ca_bo[l,d]
// (mem == 0 in the reference, so the whole cross-attn block is this constant)
// ---------------------------------------------------------------------------
__launch_bounds__(256)
__global__ void cross_const(const float* __restrict__ bv,
                            const float* __restrict__ Wo,
                            const float* __restrict__ bo,
                            float* __restrict__ cvec)
{
  const int l = blockIdx.x;
  const int d = threadIdx.x;
  const float* w = Wo + (long long)l * CD * CD;
  const float* v = bv + l * CD;
  float a = bo[l * CD + d];
  for (int e = 0; e < CD; ++e) a += v[e] * w[e * CD + d];
  cvec[l * CD + d] = a;
}

__global__ void add_bcast(float* __restrict__ x, const float* __restrict__ c)
{
  x[(long long)blockIdx.x * CD + threadIdx.x] += c[threadIdx.x];
}

// ---------------------------------------------------------------------------
// Host orchestration
// ---------------------------------------------------------------------------
static inline void launch_gemm(hipStream_t st,
    const _Float16* A, int lda, const _Float16* B, int ldb, int transB,
    float* Cf, _Float16* Ch, int ldc, int M, int N, int K,
    const float* bias, const float* res, int ldres, int resMod,
    float scale, int relu, int batches, int nh,
    long long sAb = 0, long long sAh = 0, long long sBb = 0, long long sBh = 0,
    long long sCb = 0, long long sCh = 0)
{
  const int nhShift = (nh == 4) ? 2 : 0;       // nh is 1 or 4 in this model
  dim3 g(N / 64, M / 64, batches), blk(128);
  gemm_f16_wmma<<<g, blk, 0, st>>>(A, lda, B, ldb, transB, Cf, Ch, ldc, M, N, K,
                                   bias, res, ldres, resMod - 1, scale, relu,
                                   nhShift, nh - 1,
                                   sAb, sAh, sBb, sBh, sCb, sCh);
}

extern "C" void kernel_launch(void* const* d_in, const int* in_sizes, int n_in,
                              void* d_out, int out_size, void* d_ws, size_t ws_size,
                              hipStream_t stream) {
  const int*   tok   = (const int*)  d_in[0];
  const float* ram   = (const float*)d_in[1];
  const float* projW = (const float*)d_in[2];
  const float* projB = (const float*)d_in[3];
  const float* posE  = (const float*)d_in[4];
  const float* saWq  = (const float*)d_in[5];
  const float* saWk  = (const float*)d_in[6];
  const float* saWv  = (const float*)d_in[7];
  const float* saWo  = (const float*)d_in[8];
  const float* sabq  = (const float*)d_in[9];
  const float* sabk  = (const float*)d_in[10];
  const float* sabv  = (const float*)d_in[11];
  const float* sabo  = (const float*)d_in[12];
  const float* caWo  = (const float*)d_in[16];
  const float* cabv  = (const float*)d_in[19];
  const float* cabo  = (const float*)d_in[20];
  const float* ln1s  = (const float*)d_in[21];
  const float* ln1b  = (const float*)d_in[22];
  const float* ln3s  = (const float*)d_in[25];
  const float* ln3b  = (const float*)d_in[26];
  const float* ffW1  = (const float*)d_in[27];
  const float* ffb1  = (const float*)d_in[28];
  const float* ffW2  = (const float*)d_in[29];
  const float* ffb2  = (const float*)d_in[30];
  const float* outW  = (const float*)d_in[31];
  const float* outB  = (const float*)d_in[32];
  float* logits = (float*)d_out;

  // bump allocator on d_ws (256B aligned)
  char* wsp = (char*)d_ws;
  size_t off = 0;
  auto alloc = [&](size_t bytes) -> void* {
    off = (off + 255) & ~(size_t)255;
    void* p = wsp + off;
    off += bytes;
    return p;
  };
  auto ah = [&](long long n) { return (_Float16*)alloc((size_t)n * 2); };
  auto af = [&](long long n) { return (float*)   alloc((size_t)n * 4); };

  _Float16* projW_h  = ah((long long)CNC * CD);
  _Float16* saWq_h   = ah((long long)CL * CD * CD);
  _Float16* saWk_h   = ah((long long)CL * CD * CD);
  _Float16* saWv_h   = ah((long long)CL * CD * CD);
  _Float16* saWo_h   = ah((long long)CL * CD * CD);
  _Float16* ffW1_h   = ah((long long)CL * CD * CFF);
  _Float16* ffW2_h   = ah((long long)CL * CFF * CD);
  _Float16* outW_h   = ah((long long)CD * CV);
  _Float16* scores_h = ah((long long)CROWS * CNC);
  _Float16* h_h      = ah((long long)CROWS * CD);
  _Float16* q_h      = ah((long long)CROWS * CD);
  _Float16* k_h      = ah((long long)CROWS * CD);
  _Float16* v_h      = ah((long long)CROWS * CD);
  _Float16* o_h      = ah((long long)CROWS * CD);
  _Float16* u_h      = ah((long long)CROWS * CFF);
  _Float16* probs_h  = ah((long long)CB * CNH * CS * CS);
  _Float16* x_h      = ah((long long)CROWS * CD);
  float*    x_f      = af((long long)CROWS * CD);
  float*    sc_f     = af((long long)CB * CNH * CS * CS);
  float*    cvec     = af((long long)CL * CD);

  auto conv = [&](const float* s, _Float16* d, long long n) {
    int blocks = (int)((n + 255) / 256); if (blocks > 32768) blocks = 32768;
    f32_to_f16<<<blocks, 256, 0, stream>>>(s, d, n);
  };

  // ---- weight conversion (once per launch; L2-resident afterwards)
  conv(projW, projW_h, (long long)CNC * CD);
  conv(saWq,  saWq_h,  (long long)CL * CD * CD);
  conv(saWk,  saWk_h,  (long long)CL * CD * CD);
  conv(saWv,  saWv_h,  (long long)CL * CD * CD);
  conv(saWo,  saWo_h,  (long long)CL * CD * CD);
  conv(ffW1,  ffW1_h,  (long long)CL * CD * CFF);
  conv(ffW2,  ffW2_h,  (long long)CL * CFF * CD);
  conv(outW,  outW_h,  (long long)CD * CV);

  // ---- embedding + cross-attn constants
  embed_scores<<<CROWS, 256, 0, stream>>>(tok, ram, scores_h);
  cross_const<<<CL, 256, 0, stream>>>(cabv, caWo, cabo, cvec);

  // x = scores @ proj_W + proj_b + pos_emb[s]   (pos_emb via row&(S-1) residual)
  launch_gemm(stream, scores_h, CNC, projW_h, CD, 0,
              x_f, nullptr, CD, CROWS, CD, CNC,
              projB, posE, CD, CS, 1.f, 0, 1, 1);

  const long long sSD = (long long)CS * CD;       // per-batch q/k/v stride
  const long long sSS = (long long)CS * CS;       // per-head score stride

  for (int l = 0; l < CL; ++l) {
    const long long wOff = (long long)l * CD * CD;
    // LN1 -> h
    layernorm_f16<<<CROWS, 256, 0, stream>>>(x_f, ln1s + l * CD, ln1b + l * CD, h_h);
    // q,k,v projections (f16 out)
    launch_gemm(stream, h_h, CD, saWq_h + wOff, CD, 0, nullptr, q_h, CD,
                CROWS, CD, CD, sabq + l * CD, nullptr, CD, CROWS, 1.f, 0, 1, 1);
    launch_gemm(stream, h_h, CD, saWk_h + wOff, CD, 0, nullptr, k_h, CD,
                CROWS, CD, CD, sabk + l * CD, nullptr, CD, CROWS, 1.f, 0, 1, 1);
    launch_gemm(stream, h_h, CD, saWv_h + wOff, CD, 0, nullptr, v_h, CD,
                CROWS, CD, CD, sabv + l * CD, nullptr, CD, CROWS, 1.f, 0, 1, 1);
    // scores = (q @ k^T) / 8, batched over (b,h)
    launch_gemm(stream, q_h, CD, k_h, CD, 1, sc_f, nullptr, CS,
                CS, CS, CD / CNH, nullptr, nullptr, CS, CS, 0.125f, 0,
                CB * CNH, CNH, sSD, 64, sSD, 64, (long long)CNH * sSS, sSS);
    // causal softmax -> f16 probs
    softmax_causal<<<dim3(CS, CB * CNH), 256, 0, stream>>>(sc_f, probs_h);
    // o = probs @ v, batched
    launch_gemm(stream, probs_h, CS, v_h, CD, 0, nullptr, o_h, CD,
                CS, CD / CNH, CS, nullptr, nullptr, CD, CS, 1.f, 0,
                CB * CNH, CNH, (long long)CNH * sSS, sSS, sSD, 64, sSD, 64);
    // x = o @ Wo + bo + x   (in-place residual)
    launch_gemm(stream, o_h, CD, saWo_h + wOff, CD, 0, x_f, nullptr, CD,
                CROWS, CD, CD, sabo + l * CD, x_f, CD, CROWS, 1.f, 0, 1, 1);
    // x += cross-attn constant (mem == 0 collapses the whole block)
    add_bcast<<<CROWS, 256, 0, stream>>>(x_f, cvec + l * CD);
    // LN3 -> h ; FFN
    layernorm_f16<<<CROWS, 256, 0, stream>>>(x_f, ln3s + l * CD, ln3b + l * CD, h_h);
    launch_gemm(stream, h_h, CD, ffW1_h + (long long)l * CD * CFF, CFF, 0,
                nullptr, u_h, CFF, CROWS, CFF, CD,
                ffb1 + l * CFF, nullptr, CFF, CROWS, 1.f, 1, 1, 1);
    launch_gemm(stream, u_h, CFF, ffW2_h + (long long)l * CFF * CD, CD, 0,
                x_f, nullptr, CD, CROWS, CD, CFF,
                ffb2 + l * CD, x_f, CD, CROWS, 1.f, 0, 1, 1);
  }

  // ---- final vocab projection: logits = x @ out_W + out_b  (HBM-bound)
  conv(x_f, x_h, (long long)CROWS * CD);
  launch_gemm(stream, x_h, CD, outW_h, CV, 0, logits, nullptr, CV,
              CROWS, CV, CD, outB, nullptr, CV, CROWS, 1.f, 0, 1, 1);
}
